// StaticEdgeGNN_85744727097867
// MI455X (gfx1250) — matrix operand order repaired and checked by reading, not compiled, not run on hardware
//
#include <hip/hip_runtime.h>
#include <hip/hip_bf16.h>

// ---------------------------------------------------------------------------
// Types for CDNA5 WMMA (wave32): D = A(16x32 bf16) * B(32x16 bf16) + C(16x16 f32)
// ---------------------------------------------------------------------------
typedef __attribute__((ext_vector_type(16))) __bf16          v16bf;
typedef __attribute__((ext_vector_type(16))) unsigned short  v16us;
typedef __attribute__((ext_vector_type(8)))  float           v8f;

__device__ __forceinline__ unsigned short f2bf(float f) {
    // round-to-nearest-even f32 -> bf16
    unsigned u = __float_as_uint(f);
    return (unsigned short)((u + 0x7FFFu + ((u >> 16) & 1u)) >> 16);
}

// A-fragment from an f32 row (convert in registers).
// Per ISA 05_wmma: v0..v3 hold K = kb..kb+7 (pairs), v4..v7 hold K = kb+16..kb+23.
__device__ __forceinline__ v16bf pack_a_f32(const float* __restrict__ rowp, int kb) {
    const float4* q0 = reinterpret_cast<const float4*>(rowp + kb);
    const float4* q1 = reinterpret_cast<const float4*>(rowp + kb + 16);
    float4 f0 = q0[0], f1 = q0[1], f2 = q1[0], f3 = q1[1];
    float t[16] = {f0.x,f0.y,f0.z,f0.w, f1.x,f1.y,f1.z,f1.w,
                   f2.x,f2.y,f2.z,f2.w, f3.x,f3.y,f3.z,f3.w};
    v16us u;
#pragma unroll
    for (int i = 0; i < 16; ++i) u[i] = f2bf(t[i]);
    union { v16us u; v16bf b; } cv; cv.u = u; return cv.b;
}

// A-fragment from a bf16 row: two raw 16B loads (K kb..kb+7 and kb+16..kb+23).
__device__ __forceinline__ v16bf load_a_bf16(const unsigned short* __restrict__ rowp, int kb) {
    union { uint4 q[2]; v16bf b; } cv;
    cv.q[0] = *reinterpret_cast<const uint4*>(rowp + kb);
    cv.q[1] = *reinterpret_cast<const uint4*>(rowp + kb + 16);
    return cv.b;
}

// B-fragment: lane (l&15) = col N; half selects K kb..kb+7 and kb+16..kb+23
// (in the pre-transposed weight Wt[N][K]) -> two 16B loads.
__device__ __forceinline__ v16bf load_b(const unsigned short* __restrict__ Wt, size_t off) {
    const uint4* p = reinterpret_cast<const uint4*>(Wt + off);
    union { uint4 q[2]; v16bf b; } cv;
    cv.q[0] = p[0]; cv.q[1] = p[2];
    return cv.b;
}

// ---------------------------------------------------------------------------
// Small helper kernels
// ---------------------------------------------------------------------------
__global__ void k_wt_transpose(const float* __restrict__ W, unsigned short* __restrict__ Wt,
                               int K, int N) {
    int i = blockIdx.x * blockDim.x + threadIdx.x;
    if (i < K * N) {
        int k = i / N, n = i - k * N;
        Wt[(size_t)n * K + k] = f2bf(W[i]);
    }
}

__global__ void k_deg_init(float* __restrict__ deg, int n) {
    int i = blockIdx.x * blockDim.x + threadIdx.x;
    if (i < n) deg[i] = 1.0f;                     // self loop
}

__global__ void k_deg_edge(const int* __restrict__ dst, float* __restrict__ deg, int E) {
    int e = blockIdx.x * blockDim.x + threadIdx.x;
    if (e < E) atomicAdd(&deg[dst[e]], 1.0f);     // exact small-int sums: deterministic
}

__global__ void k_dinv(const float* __restrict__ deg, float* __restrict__ dinv, int n) {
    int i = blockIdx.x * blockDim.x + threadIdx.x;
    if (i < n) dinv[i] = rsqrtf(deg[i]);
}

// agg[i,f] = xw[i,f] * dinv[i]^2   (self-loop message, also initializes agg)
__global__ void k_agg_init(const float* __restrict__ xw, const float* __restrict__ dinv,
                           float* __restrict__ agg, int total, int hshift) {
    int i = blockIdx.x * blockDim.x + threadIdx.x;
    if (i < total) {
        float di = dinv[i >> hshift];
        agg[i] = xw[i] * di * di;
    }
}

// scatter-add: agg[dst] += xw[src] * dinv[src]*dinv[dst], 4 floats per thread
__global__ void k_scatter(const float* __restrict__ xw, const float* __restrict__ dinv,
                          const int* __restrict__ src, const int* __restrict__ dst,
                          float* __restrict__ agg, int E, int H, int eshift) {
    int t = blockIdx.x * blockDim.x + threadIdx.x;
    int e = t >> eshift;
    if (e >= E) return;
    int f4 = (t & ((1 << eshift) - 1)) * 4;
    int s = src[e], d = dst[e];
    float w = dinv[s] * dinv[d];
    float4 v = *reinterpret_cast<const float4*>(xw + (size_t)s * H + f4);
    float* o = agg + (size_t)d * H + f4;
    atomicAdd(o + 0, v.x * w);
    atomicAdd(o + 1, v.y * w);
    atomicAdd(o + 2, v.z * w);
    atomicAdd(o + 3, v.w * w);
}

// h_bf16[i] = bf16(relu(agg[i] + b[f]))   — activations live in bf16 from here on
__global__ void k_bias_relu_bf16(const float* __restrict__ agg, const float* __restrict__ b,
                                 unsigned short* __restrict__ hbf, int total, int hmask) {
    int i = blockIdx.x * blockDim.x + threadIdx.x;
    if (i < total) hbf[i] = f2bf(fmaxf(agg[i] + b[i & hmask], 0.0f));
}

// ---------------------------------------------------------------------------
// Node GEMM: out[M,128] = act(A[M,K] @ W[K,128] + bias)
// Block = 256 threads = 8 waves; block owns 16 rows, wave owns a 16-col tile.
// ABF16:   A matrix is bf16 (raw fragment loads) vs f32 (convert in regs).
// RELUOUT: epilogue = bias+relu, emit bf16; else plain f32 (no bias).
// ---------------------------------------------------------------------------
template <int K, bool ABF16, bool RELUOUT>
__global__ __launch_bounds__(256) void k_gemm(const float* __restrict__ Af,
                                              const unsigned short* __restrict__ Abf,
                                              const unsigned short* __restrict__ Wt, // [128][K] bf16
                                              const float* __restrict__ bias,
                                              float* __restrict__ outf,
                                              unsigned short* __restrict__ outbf) {
    const int H = 128;
    int lane = threadIdx.x & 31;
    int wave = threadIdx.x >> 5;          // 0..7 -> N tile
    int hi   = lane >> 4;
    int r    = lane & 15;
    int mt   = blockIdx.x;

    size_t rowoff = (size_t)(mt * 16 + r) * K;
    int col = wave * 16 + r;
    __builtin_prefetch(Wt + (size_t)col * K, 0, 1);

    v8f c = {};
#pragma unroll
    for (int ks = 0; ks < K; ks += 32) {
        v16bf a;
        if constexpr (ABF16) a = load_a_bf16(Abf + rowoff, ks + hi * 8);
        else                 a = pack_a_f32(Af + rowoff, ks + hi * 8);
        v16bf b = load_b(Wt, (size_t)col * K + ks + hi * 8);
        c = __builtin_amdgcn_wmma_f32_16x16x32_bf16(false, a, false, b, (short)0, c, false, false);
    }

    size_t obase = (size_t)(mt * 16 + hi * 8) * H + col;
    if constexpr (RELUOUT) {
        float bv = bias[col];
#pragma unroll
        for (int i = 0; i < 8; ++i)        // D layout: lane l, v_i -> row i + 8*hi, col l&15
            outbf[obase + (size_t)i * H] = f2bf(fmaxf(c[i] + bv, 0.0f));
    } else {
#pragma unroll
        for (int i = 0; i < 8; ++i)
            outf[obase + (size_t)i * H] = c[i];
    }
}

// ---------------------------------------------------------------------------
// Fused edge MLP: logits[e] = relu([h[src]||h[dst]] @ Wm1 + bm1) . Wm2 + bm2
// h is bf16 [N][128]; block = 16 edges, 8 waves; wave owns a 16-col tile of hid;
// K = 256 fully unrolled (8 WMMAs), zero conversion VALU in the loop.
// ---------------------------------------------------------------------------
template <int H>   // H = 128
__global__ __launch_bounds__(256) void k_edge_mlp(const unsigned short* __restrict__ hbf,
                                                  const int* __restrict__ src,
                                                  const int* __restrict__ dst,
                                                  const unsigned short* __restrict__ WtM1, // [H][2H] bf16
                                                  const float* __restrict__ bm1,
                                                  const float* __restrict__ Wm2,
                                                  const float* __restrict__ bm2,
                                                  float* __restrict__ out) {
    __shared__ float red[8 * 16];
    int lane = threadIdx.x & 31;
    int wave = threadIdx.x >> 5;
    int hi   = lane >> 4;
    int r    = lane & 15;

    int e = blockIdx.x * 16 + r;
    int s = src[e], d = dst[e];
    const unsigned short* hs = hbf + (size_t)s * H;
    const unsigned short* hd = hbf + (size_t)d * H;
    int col = wave * 16 + r;
    __builtin_prefetch(WtM1 + (size_t)col * (2 * H), 0, 1);

    v8f c = {};
#pragma unroll
    for (int ks = 0; ks < 2 * H; ks += 32) {
        int kbA = ks + hi * 8;                       // 8-aligned: never straddles H
        const unsigned short* base = (kbA < H) ? hs : hd;
        v16bf a = load_a_bf16(base, kbA & (H - 1));
        v16bf b = load_b(WtM1, (size_t)col * (2 * H) + ks + hi * 8);
        c = __builtin_amdgcn_wmma_f32_16x16x32_bf16(false, a, false, b, (short)0, c, false, false);
    }

    float bv = bm1[col];
    float w2 = Wm2[col];
    float part[8];
#pragma unroll
    for (int i = 0; i < 8; ++i) {
        float v = fmaxf(c[i] + bv, 0.0f) * w2;       // hid * Wm2, per (row=i+8*hi, col)
#pragma unroll
        for (int off = 1; off < 16; off <<= 1)       // butterfly over the 16-lane half
            v += __shfl_xor(v, off, 32);
        part[i] = v;
    }
    if (r == 0) {
#pragma unroll
        for (int i = 0; i < 8; ++i) red[wave * 16 + hi * 8 + i] = part[i];
    }
    __syncthreads();
    if (threadIdx.x < 16) {
        float sum = bm2[0];
#pragma unroll
        for (int w = 0; w < 8; ++w) sum += red[w * 16 + threadIdx.x];
        out[blockIdx.x * 16 + threadIdx.x] = sum;
    }
}

// ---------------------------------------------------------------------------
// Host launcher
// ---------------------------------------------------------------------------
static inline size_t alup(size_t x) { return (x + 255) & ~(size_t)255; }

extern "C" void kernel_launch(void* const* d_in, const int* in_sizes, int n_in,
                              void* d_out, int out_size, void* d_ws, size_t ws_size,
                              hipStream_t stream) {
    const float* x    = (const float*)d_in[0];
    const int*   ei   = (const int*)  d_in[1];
    const float* W_in = (const float*)d_in[2];
    const float* b_in = (const float*)d_in[3];
    const float* W1   = (const float*)d_in[4];
    const float* b1   = (const float*)d_in[5];
    const float* W2   = (const float*)d_in[6];
    const float* b2   = (const float*)d_in[7];
    const float* Wm1  = (const float*)d_in[8];
    const float* bm1  = (const float*)d_in[9];
    const float* Wm2  = (const float*)d_in[10];
    const float* bm2  = (const float*)d_in[11];
    float*       out  = (float*)d_out;

    const int H  = in_sizes[3];            // 128
    const int D  = in_sizes[2] / H;        // 96
    const int Nn = in_sizes[0] / D;        // 50000
    const int E  = in_sizes[1] / 2;        // 800000

    char* w = (char*)d_ws;
    float*          bufA = (float*)w;          w += alup((size_t)Nn * H * 4);  // xw (f32)
    float*          bufB = (float*)w;          w += alup((size_t)Nn * H * 4);  // agg (f32)
    unsigned short* hbf  = (unsigned short*)w; w += alup((size_t)Nn * H * 2);  // h (bf16)
    float*          deg  = (float*)w;          w += alup((size_t)Nn * 4);
    float*          dinv = (float*)w;          w += alup((size_t)Nn * 4);
    unsigned short* WtIn = (unsigned short*)w; w += alup((size_t)D * H * 2);
    unsigned short* Wt1  = (unsigned short*)w; w += alup((size_t)H * H * 2);
    unsigned short* Wt2  = (unsigned short*)w; w += alup((size_t)H * H * 2);
    unsigned short* WtM1 = (unsigned short*)w; w += alup((size_t)2 * H * H * 2);

    const int* src = ei;
    const int* dst = ei + E;

    const int hshift = 31 - __builtin_clz(H);        // 7
    const int eshift = 31 - __builtin_clz(H / 4);    // 5
    const int total  = Nn * H;

    // Pre-transpose/convert all weights to bf16 [N][K] fragment-friendly layout.
    k_wt_transpose<<<(D * H + 255) / 256, 256, 0, stream>>>(W_in, WtIn, D, H);
    k_wt_transpose<<<(H * H + 255) / 256, 256, 0, stream>>>(W1, Wt1, H, H);
    k_wt_transpose<<<(H * H + 255) / 256, 256, 0, stream>>>(W2, Wt2, H, H);
    k_wt_transpose<<<(2 * H * H + 255) / 256, 256, 0, stream>>>(Wm1, WtM1, 2 * H, H);

    // Symmetric GCN normalization: deg = indeg(dst) + 1; dinv = rsqrt(deg).
    k_deg_init<<<(Nn + 255) / 256, 256, 0, stream>>>(deg, Nn);
    k_deg_edge<<<(E + 255) / 256, 256, 0, stream>>>(dst, deg, E);
    k_dinv<<<(Nn + 255) / 256, 256, 0, stream>>>(deg, dinv, Nn);

    // h0 = bf16(relu(x @ W_in + b_in))   (A = f32, epilogue -> bf16)
    k_gemm<96, false, true><<<Nn / 16, 256, 0, stream>>>(x, nullptr, WtIn, b_in, nullptr, hbf);

    // --- GCN layer 1 ---
    k_gemm<128, true, false><<<Nn / 16, 256, 0, stream>>>(nullptr, hbf, Wt1, nullptr, bufA, nullptr);
    k_agg_init<<<(total + 255) / 256, 256, 0, stream>>>(bufA, dinv, bufB, total, hshift);
    k_scatter<<<(E * (H / 4) + 255) / 256, 256, 0, stream>>>(bufA, dinv, src, dst, bufB, E, H, eshift);
    k_bias_relu_bf16<<<(total + 255) / 256, 256, 0, stream>>>(bufB, b1, hbf, total, H - 1);

    // --- GCN layer 2 ---
    k_gemm<128, true, false><<<Nn / 16, 256, 0, stream>>>(nullptr, hbf, Wt2, nullptr, bufA, nullptr);
    k_agg_init<<<(total + 255) / 256, 256, 0, stream>>>(bufA, dinv, bufB, total, hshift);
    k_scatter<<<(E * (H / 4) + 255) / 256, 256, 0, stream>>>(bufA, dinv, src, dst, bufB, E, H, eshift);
    k_bias_relu_bf16<<<(total + 255) / 256, 256, 0, stream>>>(bufB, b2, hbf, total, H - 1);

    // --- fused edge MLP -> logits ---
    k_edge_mlp<128><<<E / 16, 256, 0, stream>>>(hbf, src, dst, WtM1, bm1, Wm2, bm2, out);
}
